// LSTM_CRF_34540126995172
// MI455X (gfx1250) — matrix operand order repaired and checked by reading, not compiled, not run on hardware
//
#include <hip/hip_runtime.h>

// ---------------------------------------------------------------------------
// BiLSTM-CRF loss for MI455X (gfx1250, wave32, WMMA 16x16x32 f16->f32).
// ---------------------------------------------------------------------------

typedef _Float16 h8v  __attribute__((ext_vector_type(8)));
typedef _Float16 h16v __attribute__((ext_vector_type(16)));
typedef float    f8v  __attribute__((ext_vector_type(8)));

constexpr int BATCH = 128, SEQ = 256, CLEN = 16;
constexpr int WED = 100, WLD = 100, CED = 25, CLD = 25, NLAB = 9;
constexpr int WFK   = 160;  // padded word-feature K (100 emb + 50 char -> 160)
constexpr int WHK   = 224;  // padded wh width (200 -> 224)
constexpr int WKTOT = 288;  // word LSTM K: 160 input + 128 hidden (100 -> 128)
constexpr int WNG   = 448;  // word gate cols: 7 hidden tiles * 4 gates * 16
constexpr int CKTOT = 64;   // char LSTM K: 32 input (25 pad) + 32 hidden (25 pad)
constexpr int CNG   = 128;  // char gate cols: 2 hidden tiles * 4 gates * 16
constexpr float SMALLV = -1000.f;

// ---------------- WMMA helpers (CDNA5 layouts, wave32) ----------------------

__device__ __forceinline__ f8v wmma_f16(h16v a, h16v b, f8v c) {
  // v_wmma_f32_16x16x32_f16: 8-arg form (neg_a, A, neg_b, B, c_mod, C, reuse_a, reuse_b)
  return __builtin_amdgcn_wmma_f32_16x16x32_f16(false, a, false, b, (short)0, c,
                                                false, false);
}

__device__ __forceinline__ h16v join16(h8v lo, h8v hi) {
  return __builtin_shufflevector(lo, hi, 0, 1, 2, 3, 4, 5, 6, 7, 8, 9, 10, 11,
                                 12, 13, 14, 15);
}

// A fragment (16x32 f16). ISA layout: lane = (half<<4)|m ; element i holds
// K = kbase + (i<8 ? 8*half + i : 16 + 8*half + (i-8)).  Source: LDS row-major.
__device__ __forceinline__ h16v load_a_frag(const _Float16* base, int kbase,
                                            int stride) {
  int lane = threadIdx.x & 31;
  int m = lane & 15, hv = lane >> 4;
  const _Float16* p = base + m * stride + kbase;
  h8v lo = *(const h8v*)(p + 8 * hv);
  h8v hi = *(const h8v*)(p + 16 + 8 * hv);
  return join16(lo, hi);
}

// B fragment (32x16 f16). Lane holds output column n = nbase + lane%16,
// element i holds K = kbase + (i<8 ? 8*half + i : 16 + 8*half + (i-8)).  Weights stored [N][Kstride]
// f16 so each lane reads two non-adjacent 16-byte runs (two b128 loads, L2-resident).
__device__ __forceinline__ h16v load_b_frag(const _Float16* w, int nbase,
                                            int kbase, int kstride) {
  int lane = threadIdx.x & 31;
  int n = nbase + (lane & 15), hv = lane >> 4;
  const _Float16* p = w + (size_t)n * kstride + kbase;
  h8v lo = *(const h8v*)(p + 8 * hv);
  h8v hi = *(const h8v*)(p + 16 + 8 * hv);
  return join16(lo, hi);
}

__device__ __forceinline__ float sigm(float x) { return 1.f / (1.f + __expf(-x)); }
__device__ __forceinline__ f8v splat8(float x) {
  f8v v = {x, x, x, x, x, x, x, x};
  return v;
}

// ---------------- Weight preparation (reorder + transpose + pad, f32->f16) --
// Gate columns reordered as n = q*64 + g*16 + u  (hidden tile q, gate g in
// i,f,g,o order, unit u).  Padded units/K are zero => recurrence stays exact.

__global__ void prep_char_kernel(const float* WihF, const float* WhhF,
                                 const float* bF, const float* WihB,
                                 const float* WhhB, const float* bB,
                                 _Float16* Wc, float* bc) {
  int i = blockIdx.x * blockDim.x + threadIdx.x;
  if (i >= 2 * CNG * CKTOT) return;
  int d = i / (CNG * CKTOT), rem = i % (CNG * CKTOT);
  int n = rem / CKTOT, k = rem % CKTOT;
  int q = n >> 6, g = (n >> 4) & 3, u = n & 15, j = q * 16 + u;
  bool valid = j < CLD;
  int r = g * CLD + j;
  const float* Wih = d ? WihB : WihF;
  const float* Whh = d ? WhhB : WhhF;
  const float* bb = d ? bB : bF;
  float v = 0.f;
  if (valid) {
    if (k < 32) { if (k < CED) v = Wih[r * CED + k]; }
    else        { int kk = k - 32; if (kk < CLD) v = Whh[r * CLD + kk]; }
  }
  Wc[i] = (_Float16)v;
  if (k == 0) bc[d * CNG + n] = valid ? bb[r] : 0.f;
}

__global__ void prep_word_kernel(const float* WihF, const float* WhhF,
                                 const float* bF, const float* WihB,
                                 const float* WhhB, const float* bB,
                                 _Float16* Ww, float* bw) {
  int i = blockIdx.x * blockDim.x + threadIdx.x;
  if (i >= 2 * WNG * WKTOT) return;
  int d = i / (WNG * WKTOT), rem = i % (WNG * WKTOT);
  int n = rem / WKTOT, k = rem % WKTOT;
  int q = n / 64, g = (n >> 4) & 3, u = n & 15, j = q * 16 + u;
  bool valid = j < WLD;
  int r = g * WLD + j;
  const float* Wih = d ? WihB : WihF;
  const float* Whh = d ? WhhB : WhhF;
  const float* bb = d ? bB : bF;
  float v = 0.f;
  if (valid) {
    if (k < 160) { if (k < WED + 2 * CLD) v = Wih[r * (WED + 2 * CLD) + k]; }
    else         { int kk = k - 160; if (kk < WLD) v = Whh[r * WLD + kk]; }
  }
  Ww[i] = (_Float16)v;
  if (k == 0) bw[d * WNG + n] = valid ? bb[r] : 0.f;
}

__global__ void prep_cls_kernel(const float* W1, const float* b1,
                                const float* W2, const float* b2, _Float16* W1p,
                                float* b1p, _Float16* W2p, float* b2p) {
  int i = blockIdx.x * blockDim.x + threadIdx.x;
  const int T1 = 112 * WHK, T2 = 16 * 128;
  if (i < T1) {
    int n = i / WHK, k = i % WHK;
    float v = (n < WLD && k < 2 * WLD) ? W1[n * (2 * WLD) + k] : 0.f;
    W1p[i] = (_Float16)v;
  } else if (i < T1 + T2) {
    int j = i - T1, n = j / 128, k = j % 128;
    float v = (n < NLAB && k < WLD) ? W2[n * WLD + k] : 0.f;
    W2p[j] = (_Float16)v;
  } else if (i < T1 + T2 + 112) {
    int n = i - T1 - T2;
    b1p[n] = (n < WLD) ? b1[n] : 0.f;
  } else if (i < T1 + T2 + 112 + 16) {
    int n = i - T1 - T2 - 112;
    b2p[n] = (n < NLAB) ? b2[n] : 0.f;
  }
}

// ---------------- Feature assembly ------------------------------------------

__global__ void build_wf_kernel(const float* word_emb, const int* word_ids,
                                _Float16* wf) {
  size_t i = (size_t)blockIdx.x * blockDim.x + threadIdx.x;
  size_t total = (size_t)BATCH * SEQ * WFK;
  if (i >= total) return;
  int k = (int)(i % WFK);
  size_t row = i / WFK;
  float v = 0.f;
  if (k < WED) v = word_emb[(size_t)word_ids[row] * WED + k];
  wf[i] = (_Float16)v;  // char cols overwritten by char kernel; 150..159 pad 0
}

__global__ void zero_wh_kernel(_Float16* wh) {
  size_t i = (size_t)blockIdx.x * blockDim.x + threadIdx.x;
  if (i < (size_t)BATCH * SEQ * WHK) wh[i] = (_Float16)0;
}

// ---------------- Char BiLSTM: one wave per 16 sequences --------------------

__global__ void char_lstm_kernel(const float* char_emb, const int* char_ids,
                                 const int* word_num, const _Float16* Wc,
                                 const float* bc, _Float16* wf) {
  __shared__ _Float16 arow[4][16][CKTOT];  // [x | h] rows per wave
  int wave = threadIdx.x >> 5, lane = threadIdx.x & 31;
  int seq0 = (blockIdx.x * 4 + wave) * 16;
  int u = lane & 15, hv = lane >> 4;
  for (int dir = 0; dir < 2; ++dir) {
    const _Float16* W = Wc + dir * CNG * CKTOT;
    const float* bb = bc + dir * CNG;
    for (int idx = lane; idx < 16 * 32; idx += 32)
      arow[wave][idx >> 5][32 + (idx & 31)] = (_Float16)0;
    f8v c0 = splat8(0.f), c1 = splat8(0.f);
    __syncthreads();
    for (int stp = 0; stp < CLEN; ++stp) {
      int t = dir ? (CLEN - 1 - stp) : stp;
      for (int idx = lane; idx < 16 * 32; idx += 32) {
        int m = idx >> 5, k = idx & 31;
        int cid = char_ids[(size_t)(seq0 + m) * CLEN + t];
        arow[wave][m][k] =
            (k < CED) ? (_Float16)char_emb[(size_t)cid * CED + k] : (_Float16)0;
      }
      __syncthreads();
      h16v a0 = load_a_frag(&arow[wave][0][0], 0, CKTOT);
      h16v a1 = load_a_frag(&arow[wave][0][0], 32, CKTOT);
      f8v acc[8];
#pragma unroll
      for (int q = 0; q < 2; ++q) {
#pragma unroll
        for (int g = 0; g < 4; ++g) {
          int nb = (q * 4 + g) * 16;
          f8v a_ = splat8(bb[nb + u]);
          a_ = wmma_f16(a0, load_b_frag(W, nb, 0, CKTOT), a_);
          a_ = wmma_f16(a1, load_b_frag(W, nb, 32, CKTOT), a_);
          acc[q * 4 + g] = a_;
        }
      }
      __syncthreads();
#pragma unroll
      for (int q = 0; q < 2; ++q) {
        f8v cprev = q ? c1 : c0, cnew;
#pragma unroll
        for (int v = 0; v < 8; ++v) {
          float cv = sigm(acc[q * 4 + 1][v]) * cprev[v] +
                     sigm(acc[q * 4 + 0][v]) * tanhf(acc[q * 4 + 2][v]);
          float hval = sigm(acc[q * 4 + 3][v]) * tanhf(cv);
          cnew[v] = cv;
          arow[wave][v + 8 * hv][32 + q * 16 + u] = (_Float16)hval;
        }
        if (q) c1 = cnew; else c0 = cnew;
      }
      __syncthreads();
    }
    // masked writeback of final hidden -> wf char columns
    for (int idx = lane; idx < 16 * CLD; idx += 32) {
      int m = idx / CLD, j = idx % CLD;
      int row = seq0 + m, b = row / SEQ, s = row % SEQ;
      float mask = (s < word_num[b]) ? 1.f : 0.f;
      float h = (float)arow[wave][m][32 + j];
      wf[(size_t)row * WFK + WED + dir * CLD + j] = (_Float16)(h * mask);
    }
    __syncthreads();
  }
}

// ---------------- Word BiLSTM: workgroup per (16 rows, direction) -----------

__global__ void word_lstm_kernel(const _Float16* wf, const _Float16* Ww_all,
                                 const float* bw_all, _Float16* wh) {
  __shared__ _Float16 arow[16][WKTOT];  // cols 0..159 = x_t, 160..287 = h
  int tid = threadIdx.x, wave = tid >> 5, lane = tid & 31;
  int dir = blockIdx.x & 1, b0 = (blockIdx.x >> 1) * 16;
  int u = lane & 15, hv = lane >> 4;
  const _Float16* W = Ww_all + (size_t)dir * WNG * WKTOT;
  const float* bb = bw_all + dir * WNG;
  for (int idx = tid; idx < 16 * 128; idx += 224)
    arow[idx >> 7][160 + (idx & 127)] = (_Float16)0;
  f8v cacc = splat8(0.f);
  __syncthreads();
  int q = wave;  // hidden tile owned by this wave (0..6)
  for (int stp = 0; stp < SEQ; ++stp) {
    int t = dir ? (SEQ - 1 - stp) : stp;
    for (int idx = tid; idx < 16 * WFK; idx += 224) {
      int m = idx / WFK, k = idx - m * WFK;
      arow[m][k] = wf[((size_t)(b0 + m) * SEQ + t) * WFK + k];
    }
    __syncthreads();
    h16v af[9];
#pragma unroll
    for (int kc = 0; kc < 9; ++kc)
      af[kc] = load_a_frag(&arow[0][0], kc * 32, WKTOT);
    f8v acc[4];
#pragma unroll
    for (int g = 0; g < 4; ++g) {
      int nb = (q * 4 + g) * 16;
      f8v a_ = splat8(bb[nb + u]);
#pragma unroll
      for (int kc = 0; kc < 9; ++kc)
        a_ = wmma_f16(af[kc], load_b_frag(W, nb, kc * 32, WKTOT), a_);
      acc[g] = a_;
    }
    __syncthreads();  // everyone finished reading h before rewrites
    int j = q * 16 + u;
    f8v cnew;
#pragma unroll
    for (int v = 0; v < 8; ++v) {
      float cv = sigm(acc[1][v]) * cacc[v] + sigm(acc[0][v]) * tanhf(acc[2][v]);
      float hval = sigm(acc[3][v]) * tanhf(cv);
      cnew[v] = cv;
      int m = v + 8 * hv;
      arow[m][160 + j] = (_Float16)hval;
      if (j < WLD)
        wh[((size_t)(b0 + m) * SEQ + t) * WHK + dir * WLD + j] = (_Float16)hval;
    }
    cacc = cnew;
    __syncthreads();
  }
}

// ---------------- Classifier: fused tanh-MLP, one wave per 16 rows ----------

__global__ void classifier_kernel(const _Float16* wh, const _Float16* W1p,
                                  const float* b1p, const _Float16* W2p,
                                  const float* b2p, const int* word_num,
                                  float* feats) {
  __shared__ _Float16 awh[4][16][WHK];
  __shared__ _Float16 t1[4][16][128];
  int wave = threadIdx.x >> 5, lane = threadIdx.x & 31;
  int row0 = (blockIdx.x * 4 + wave) * 16;
  int u = lane & 15, hv = lane >> 4;
  for (int idx = lane; idx < 16 * WHK; idx += 32) {
    int m = idx / WHK, k = idx - m * WHK;
    awh[wave][m][k] = wh[(size_t)(row0 + m) * WHK + k];
  }
  for (int idx = lane; idx < 16 * 16; idx += 32)
    t1[wave][idx >> 4][112 + (idx & 15)] = (_Float16)0;
  __syncthreads();
  h16v af[7];
#pragma unroll
  for (int kc = 0; kc < 7; ++kc)
    af[kc] = load_a_frag(&awh[wave][0][0], kc * 32, WHK);
#pragma unroll
  for (int nt = 0; nt < 7; ++nt) {
    int nb = nt * 16;
    f8v a_ = splat8(b1p[nb + u]);
#pragma unroll
    for (int kc = 0; kc < 7; ++kc)
      a_ = wmma_f16(af[kc], load_b_frag(W1p, nb, kc * 32, WHK), a_);
#pragma unroll
    for (int v = 0; v < 8; ++v)
      t1[wave][v + 8 * hv][nb + u] = (_Float16)tanhf(a_[v]);
  }
  __syncthreads();
  h16v tf[4];
#pragma unroll
  for (int kc = 0; kc < 4; ++kc)
    tf[kc] = load_a_frag(&t1[wave][0][0], kc * 32, 128);
  f8v a2 = splat8(b2p[u]);
#pragma unroll
  for (int kc = 0; kc < 4; ++kc)
    a2 = wmma_f16(tf[kc], load_b_frag(W2p, 0, kc * 32, 128), a2);
#pragma unroll
  for (int v = 0; v < 8; ++v) {
    int m = v + 8 * hv, row = row0 + m, b = row / SEQ, s = row - b * SEQ;
    float mask = (s < word_num[b]) ? 1.f : 0.f;
    feats[(size_t)row * 16 + u] = a2[v] * mask;  // cols 9..15 are exact zeros
  }
}

// ---------------- CRF loss: scalar per batch row + reduction ----------------

__global__ void crf_kernel(const float* feats, const float* trans,
                           const int* word_num, const int* label_ids,
                           float* out) {
  __shared__ float tr[121];
  __shared__ float red[BATCH];
  int b = threadIdx.x;
  if (b < 121) tr[b] = trans[b];
  __syncthreads();
  int wn = word_num[b];
  wn = wn < 1 ? 1 : (wn > SEQ ? SEQ : wn);
  const int* lab = label_ids + (size_t)b * SEQ;
  const float* fb = feats + (size_t)b * SEQ * 16;
  // numerator: gold path (feats already masked)
  float num = 0.f;
  int prev = NLAB;  // start state = 9
  for (int t = 0; t < wn; ++t) {
    int l = lab[t];
    num += fb[t * 16 + l] + tr[prev * 11 + l];
    prev = l;
  }
  num += tr[prev * 11 + 10];  // -> end state
  // denominator: forward algorithm (obs row wn+1 contributes only trans[:,10])
  float alpha[11];
  for (int i = 0; i < 11; ++i) alpha[i] = (i == NLAB) ? 0.f : SMALLV;
  for (int t = 0; t < wn; ++t) {
    float na[11];
    for (int j = 0; j < 11; ++j) {
      float mx = -3.4e38f;
      for (int i = 0; i < 11; ++i) mx = fmaxf(mx, alpha[i] + tr[i * 11 + j]);
      float sm = 0.f;
      for (int i = 0; i < 11; ++i) sm += __expf(alpha[i] + tr[i * 11 + j] - mx);
      float obs = (j < NLAB) ? fb[t * 16 + j] : SMALLV;
      na[j] = obs + mx + __logf(sm);
    }
    for (int j = 0; j < 11; ++j) alpha[j] = na[j];
  }
  float mx = -3.4e38f;
  for (int i = 0; i < 11; ++i) mx = fmaxf(mx, alpha[i] + tr[i * 11 + 10]);
  float sm = 0.f;
  for (int i = 0; i < 11; ++i) sm += __expf(alpha[i] + tr[i * 11 + 10] - mx);
  float denom = mx + __logf(sm);
  red[b] = denom - num;
  __syncthreads();
  for (int s = 64; s > 0; s >>= 1) {
    if (b < s) red[b] += red[b + s];
    __syncthreads();
  }
  if (b == 0) out[0] = red[0] / (float)BATCH;
}

// ---------------- Host-side orchestration -----------------------------------

extern "C" void kernel_launch(void* const* d_in, const int* in_sizes, int n_in,
                              void* d_out, int out_size, void* d_ws,
                              size_t ws_size, hipStream_t stream) {
  const float* word_emb = (const float*)d_in[0];
  const float* char_emb = (const float*)d_in[1];
  const float* cWihF = (const float*)d_in[2];
  const float* cWhhF = (const float*)d_in[3];
  const float* cbF = (const float*)d_in[4];
  const float* cWihB = (const float*)d_in[5];
  const float* cWhhB = (const float*)d_in[6];
  const float* cbB = (const float*)d_in[7];
  const float* wWihF = (const float*)d_in[8];
  const float* wWhhF = (const float*)d_in[9];
  const float* wbF = (const float*)d_in[10];
  const float* wWihB = (const float*)d_in[11];
  const float* wWhhB = (const float*)d_in[12];
  const float* wbB = (const float*)d_in[13];
  const float* W1 = (const float*)d_in[14];
  const float* b1 = (const float*)d_in[15];
  const float* W2 = (const float*)d_in[16];
  const float* b2 = (const float*)d_in[17];
  const float* trans = (const float*)d_in[18];
  const int* word_num = (const int*)d_in[19];
  const int* word_ids = (const int*)d_in[20];
  const int* char_ids = (const int*)d_in[21];
  const int* label_ids = (const int*)d_in[22];
  float* out = (float*)d_out;

  // workspace carve-out (deterministic, 256B aligned)
  char* ws = (char*)d_ws;
  size_t off = 0;
  auto carve = [&](size_t bytes) -> void* {
    void* p = ws + off;
    off = (off + bytes + 255) & ~(size_t)255;
    return p;
  };
  _Float16* Wc = (_Float16*)carve(2 * CNG * CKTOT * sizeof(_Float16));
  float* bc = (float*)carve(2 * CNG * sizeof(float));
  _Float16* Ww = (_Float16*)carve((size_t)2 * WNG * WKTOT * sizeof(_Float16));
  float* bw = (float*)carve(2 * WNG * sizeof(float));
  _Float16* W1p = (_Float16*)carve(112 * WHK * sizeof(_Float16));
  float* b1p = (float*)carve(112 * sizeof(float));
  _Float16* W2p = (_Float16*)carve(16 * 128 * sizeof(_Float16));
  float* b2p = (float*)carve(16 * sizeof(float));
  _Float16* wf = (_Float16*)carve((size_t)BATCH * SEQ * WFK * sizeof(_Float16));
  _Float16* wh = (_Float16*)carve((size_t)BATCH * SEQ * WHK * sizeof(_Float16));
  float* feats = (float*)carve((size_t)BATCH * SEQ * 16 * sizeof(float));
  (void)in_sizes; (void)n_in; (void)out_size; (void)ws_size;

  // 1. weight prep
  prep_char_kernel<<<(2 * CNG * CKTOT + 255) / 256, 256, 0, stream>>>(
      cWihF, cWhhF, cbF, cWihB, cWhhB, cbB, Wc, bc);
  prep_word_kernel<<<(2 * WNG * WKTOT + 255) / 256, 256, 0, stream>>>(
      wWihF, wWhhF, wbF, wWihB, wWhhB, wbB, Ww, bw);
  prep_cls_kernel<<<(112 * WHK + 16 * 128 + 112 + 16 + 255) / 256, 256, 0,
                    stream>>>(W1, b1, W2, b2, W1p, b1p, W2p, b2p);
  // 2. feature buffers
  {
    size_t n = (size_t)BATCH * SEQ * WFK;
    build_wf_kernel<<<(unsigned)((n + 255) / 256), 256, 0, stream>>>(word_emb,
                                                                     word_ids,
                                                                     wf);
    size_t nz = (size_t)BATCH * SEQ * WHK;
    zero_wh_kernel<<<(unsigned)((nz + 255) / 256), 256, 0, stream>>>(wh);
  }
  // 3. char BiLSTM: 2048 tiles of 16 seqs, 4 waves/block
  char_lstm_kernel<<<512, 128, 0, stream>>>(char_emb, char_ids, word_num, Wc,
                                            bc, wf);
  // 4. word BiLSTM: 8 M-tiles x 2 directions, 7 waves/block
  word_lstm_kernel<<<16, 224, 0, stream>>>(wf, Ww, bw, wh);
  // 5. classifier: 2048 tiles, 4 waves/block
  classifier_kernel<<<512, 128, 0, stream>>>(wh, W1p, b1p, W2p, b2p, word_num,
                                             feats);
  // 6. CRF + mean reduction
  crf_kernel<<<1, BATCH, 0, stream>>>(feats, trans, word_num, label_ids, out);
}
